// inceptionSNN_64742337019935
// MI455X (gfx1250) — matrix operand order repaired and checked by reading, not compiled, not run on hardware
//
#include <hip/hip_runtime.h>

// CDNA5 / gfx1250, wave32.
typedef __attribute__((ext_vector_type(16))) _Float16 v16h;
typedef __attribute__((ext_vector_type(8)))  _Float16 v8h;
typedef __attribute__((ext_vector_type(8)))  float    v8f;
typedef __attribute__((ext_vector_type(4)))  unsigned int v4u;
typedef __attribute__((ext_vector_type(8)))  int      v8i_;
typedef __attribute__((ext_vector_type(4)))  int      v4i_;

#define TSTEPS 15

#if defined(__has_builtin)
#if __has_builtin(__builtin_amdgcn_tensor_load_to_lds) && \
    __has_builtin(__builtin_amdgcn_s_wait_tensorcnt)
#define HAVE_TDM 1
#endif
#endif
#ifndef HAVE_TDM
#define HAVE_TDM 0
#endif

// ------------------------------------------------------------------
// elementwise helpers
// ------------------------------------------------------------------
__global__ void zero_f32_kernel(float* __restrict__ p, size_t n) {
  size_t i = (size_t)blockIdx.x * blockDim.x + threadIdx.x;
  size_t stride = (size_t)gridDim.x * blockDim.x;
  for (; i < n; i += stride) p[i] = 0.0f;
}

__global__ void f32_to_f16_kernel(const float* __restrict__ in,
                                  _Float16* __restrict__ out, size_t n) {
  size_t i = (size_t)blockIdx.x * blockDim.x + threadIdx.x;
  size_t stride = (size_t)gridDim.x * blockDim.x;
  for (; i < n; i += stride) out[i] = (_Float16)in[i];
}

// Relayout weights [F][C][KH][KW] f32 -> [KH*KW][Fpad][Cpad] f16 (zero padded)
// so each (kh,kw, 64-feature, 32-channel) A-tile is a dense 2D tile for TDM.
__global__ void relayout_w_kernel(const float* __restrict__ in,
                                  _Float16* __restrict__ out,
                                  int F, int C, int KHKW, int Fpad, int Cpad) {
  size_t n = (size_t)KHKW * Fpad * Cpad;
  size_t i = (size_t)blockIdx.x * blockDim.x + threadIdx.x;
  size_t stride = (size_t)gridDim.x * blockDim.x;
  for (; i < n; i += stride) {
    int c = (int)(i % Cpad);
    int f = (int)((i / Cpad) % Fpad);
    int s = (int)(i / ((size_t)Cpad * Fpad));
    float v = (f < F && c < C) ? in[((size_t)f * C + c) * KHKW + s] : 0.0f;
    out[i] = (_Float16)v;
  }
}

// 2x2 stride-2 max pool: in [TC,IH,IW] -> out [TC,IH/2,IW/2]
__global__ void pool2x2_kernel(const _Float16* __restrict__ in,
                               _Float16* __restrict__ out,
                               int TC, int IH, int IW) {
  int OH = IH >> 1, OW = IW >> 1;
  size_t n = (size_t)TC * OH * OW;
  size_t i = (size_t)blockIdx.x * blockDim.x + threadIdx.x;
  size_t stride = (size_t)gridDim.x * blockDim.x;
  for (; i < n; i += stride) {
    int ow = (int)(i % OW);
    int oh = (int)((i / OW) % OH);
    int tc = (int)(i / ((size_t)OW * OH));
    size_t base = ((size_t)tc * IH + 2 * oh) * IW + 2 * ow;
    float m = (float)in[base];
    float v;
    v = (float)in[base + 1];      m = v > m ? v : m;
    v = (float)in[base + IW];     m = v > m ? v : m;
    v = (float)in[base + IW + 1]; m = v > m ? v : m;
    out[i] = (_Float16)m;
  }
}

// 3x3 stride-3 max pool + zero pad 2: in [TC,128,128] -> out [TC,46,46]
__global__ void pool3pad2_kernel(const _Float16* __restrict__ in,
                                 _Float16* __restrict__ out, int TC) {
  size_t n = (size_t)TC * 46 * 46;
  size_t i = (size_t)blockIdx.x * blockDim.x + threadIdx.x;
  size_t stride = (size_t)gridDim.x * blockDim.x;
  for (; i < n; i += stride) {
    int x = (int)(i % 46);
    int y = (int)((i / 46) % 46);
    int tc = (int)(i / 2116);
    float m = 0.0f;
    if (y >= 2 && y < 44 && x >= 2 && x < 44) {
      int y0 = 3 * (y - 2), x0 = 3 * (x - 2);
      #pragma unroll
      for (int dy = 0; dy < 3; ++dy)
        #pragma unroll
        for (int dx = 0; dx < 3; ++dx) {
          float v = (float)in[((size_t)tc * 128 + y0 + dy) * 128 + x0 + dx];
          m = v > m ? v : m;
        }
    }
    out[i] = (_Float16)m;
  }
}

// ------------------------------------------------------------------
// TDM: stage a [64 rows x 32 ch] f16 weight tile (row stride = Cpad) to LDS.
// D# per CDNA5 ISA ch8: group0 {count=1, lds_addr, global_addr, type=2},
// group1 {data_size=2B, tensor_dim0=Cpad, tensor_dim1=Fpad, tile=32x64,
//         tensor_dim0_stride=Cpad}, groups 2/3/4 zero (2D tensor).
// This toolchain's builtin is the 6-arg clang-23 form:
//   (uint32x4, int32x8, int32x4, int32x4, int32x8, i32 cpol)
// ------------------------------------------------------------------
#if HAVE_TDM
__device__ inline void tdm_load_a_tile(unsigned ldsAddr, const _Float16* gptr,
                                       int Cpad, int Fpad) {
  unsigned long long ga = (unsigned long long)(uintptr_t)gptr;
  v4u g0 = {0u, 0u, 0u, 0u};
  g0[0] = 1u;                                         // count=1 (valid D#)
  g0[1] = ldsAddr;                                    // lds_addr [63:32]
  g0[2] = (unsigned)(ga & 0xFFFFFFFFull);             // global_addr lo
  g0[3] = (unsigned)((ga >> 32) & 0x01FFFFFFull)      // global_addr hi (57b)
          | (2u << 30);                               // type=2 ("image")
  v8i_ g1 = {0, 0, 0, 0, 0, 0, 0, 0};
  g1[0] = (int)(1u << 16);                            // data_size=1 -> 2 bytes
  g1[1] = (int)(((unsigned)Cpad & 0xFFFFu) << 16);    // tensor_dim0 lo16 @48
  g1[2] = (int)((((unsigned)Cpad >> 16) & 0xFFFFu)    // tensor_dim0 hi16 @64
                | (((unsigned)Fpad & 0xFFFFu) << 16));// tensor_dim1 lo16 @80
  g1[3] = (int)((((unsigned)Fpad >> 16) & 0xFFFFu)    // tensor_dim1 hi16 @96
                | (32u << 16));                       // tile_dim0=32 @112
  g1[4] = 64;                                         // tile_dim1=64 @128
  g1[5] = (int)(unsigned)Cpad;                        // tensor_dim0_stride @160
  v4i_ z4 = {0, 0, 0, 0};
  v8i_ z8 = {0, 0, 0, 0, 0, 0, 0, 0};
  __builtin_amdgcn_tensor_load_to_lds(g0, g1, z4, z4, z8, 0);
}
#endif

// ------------------------------------------------------------------
// Implicit-GEMM conv on V_WMMA_F32_16X16X32_F16, channel-chunked K.
// Block = 128 threads (4 waves) = 64 features x 16 pixels.
//  - per K-chunk (32 channels at fixed kh,kw):
//      * all threads stage the 32x16 B tile into LDS in fragment order
//      * wave 0 TDM-loads the 64x32 A tile into LDS, s_wait_tensorcnt
//      * each wave: 2x ds_load_b128 per fragment, 1x v_wmma
// Template geometry -> all index math folds to constants (no div/mod).
// Fragment layouts per ISA 7.12.2 (wave32):
//  A 16x32: lane<16 -> M=lane, Kbase 0 ; lane>=16 -> M=lane-16, Kbase 8
//           elems: K = Kbase + {0..7} and Kbase + {16..23}  (two 16B runs)
//  B 32x16: lane<16 -> K=0..7,16..23 ; lane>=16 -> K=8..15,24..31 ; N = lane&15
//  C/D 16x16 f32: N = lane&15 ; VGPR r -> M = r + (lane<16 ? 0 : 8)
// ------------------------------------------------------------------
template <int C, int KH, int KW, int PAD, int MODE, int CPAD, int FPAD>
__global__ __launch_bounds__(128)
void conv_wmma(const _Float16* __restrict__ Wr,   // [KH*KW][FPAD][CPAD] f16
               const _Float16* __restrict__ X,    // [T][C][IH][IW] f16
               int F, int IH, int IW, int OH, int OW,
               float thr,
               _Float16* __restrict__ spkH,       // MODE 0 out [T,F,OH,OW]
               float* __restrict__ potOut,        // MODE 1: pot base (d_out)
               float* __restrict__ spkOut,        // MODE 1: spk base (d_out)
               int chOff, int pb) {
  constexpr int KHKW = KH * KW;
  constexpr int NC32 = (C + 31) / 32;
  constexpr int NCHUNK = KHKW * NC32;

  __shared__ _Float16 ldsA[64 * 32];   // A tile: [row][channel]
  __shared__ _Float16 ldsB[32 * 16];   // B tile in fragment order: [lane][elem]

  const int tid  = threadIdx.x;
  const int lane = tid & 31;
  const int wv   = tid >> 5;
  const int t    = blockIdx.z;
  const int nPix = OH * OW;

  // ---- B staging coordinates (fixed per thread) ----
  const int sn  = tid & 15;        // pixel within tile
  const int skq = tid >> 4;        // 0..7 -> stages channels skq*4 .. +3
  int sp = blockIdx.x * 16 + sn;
  if (sp >= nPix) sp = nPix - 1;   // clamp; garbage columns masked at writeout
  const int soy = sp / OW;
  const int sox = sp - soy * OW;
  const _Float16* __restrict__ Xt = X + (size_t)t * C * IH * IW;

  // ---- fragment coordinates ----
  const int mA = lane & 15;
  const int kABase = (lane < 16) ? 0 : 8;
  const int nB = lane & 15;
  const int p = blockIdx.x * 16 + nB;
  const bool pValid = (p < nPix);
  const int rowA = wv * 16 + mA;                 // 0..63 within feature group

  v8f acc = {};
  for (int ch = 0; ch < NCHUNK; ++ch) {
    const int ci = ch % NC32;                    // constant divisor
    const int s  = ch / NC32;
    const int kh = s / KW;                       // constant divisor
    const int kw = s - kh * KW;
    const int c0 = ci * 32;

    // stage B chunk: value X[t][c0+k][soy+kh-PAD][sox+kw-PAD]
    {
      const int iy = soy + kh - PAD;
      const int ix = sox + kw - PAD;
      const bool inb = (PAD == 0) ||
                       (iy >= 0 && iy < IH && ix >= 0 && ix < IW);
      #pragma unroll
      for (int j = 0; j < 4; ++j) {
        const int k = skq * 4 + j;
        const int c = c0 + k;
        _Float16 v = (_Float16)0.0f;
        if (inb && c < C)
          v = Xt[((size_t)c * IH + iy) * IW + ix];
        const int dl = (((k >> 3) & 1) << 4) + sn;
        ldsB[dl * 16 + ((k & 7) | ((k >> 4) << 3))] = v;
      }
    }

    // stage A tile [64 x 32] from relaid weights
#if HAVE_TDM
    if (wv == 0) {
      const _Float16* gsrc =
          Wr + ((size_t)(s * FPAD + (int)blockIdx.y * 64) * CPAD + c0);
      tdm_load_a_tile((unsigned)(uintptr_t)ldsA, gsrc, CPAD, FPAD);
      __builtin_amdgcn_s_wait_tensorcnt(0);
    }
#else
    {
      const _Float16* gsrc =
          Wr + ((size_t)(s * FPAD + (int)blockIdx.y * 64) * CPAD + c0);
      #pragma unroll
      for (int j = 0; j < 16; ++j) {
        const int idx = tid + j * 128;           // 2048 = 64*32 elements
        const int row = idx >> 5, cc = idx & 31;
        ldsA[idx] = gsrc[(size_t)row * CPAD + cc];
      }
    }
#endif
    __syncthreads();

    // fragments from LDS (two 16B ds loads each) + WMMA
    v16h a, b;
    {
      const _Float16* rp = ldsA + rowA * 32;
      const v8h lo = *(const v8h*)(rp + kABase);        // K = kABase+0..7
      const v8h hi = *(const v8h*)(rp + kABase + 16);   // K = kABase+16..23
      #pragma unroll
      for (int e = 0; e < 8; ++e) { a[e] = lo[e]; a[8 + e] = hi[e]; }
    }
    b = *(const v16h*)(ldsB + lane * 16);
    acc = __builtin_amdgcn_wmma_f32_16x16x32_f16(false, a, false, b,
                                                 (short)0, acc, false, false);
    __syncthreads();
  }

  // ---- epilogue ----
  const int mBase = (lane < 16) ? 0 : 8;
  #pragma unroll
  for (int r = 0; r < 8; ++r) {
    const int f = (int)blockIdx.y * 64 + wv * 16 + mBase + r;
    if (f >= F || !pValid) continue;
    const float pot = acc[r];
    if (MODE == 0) {
      // fire w/ threshold: spike = (pot > thr)
      spkH[((size_t)t * F + f) * nPix + p] =
          (pot > thr) ? (_Float16)1.0f : (_Float16)0.0f;
    } else {
      const int oy = p / OW, ox = p - (p / OW) * OW;
      const size_t o = (((size_t)t * 600 + chOff + f) * 46 +
                        (size_t)(pb + oy)) * 46 + (size_t)(pb + ox);
      potOut[o] = pot;
      if (t == TSTEPS - 1)   // fire_last: only last timestep spikes
        spkOut[o] = (pot > 0.0f) ? 1.0f : ((pot < 0.0f) ? -1.0f : 0.0f);
    }
  }
}

// ------------------------------------------------------------------
// winner-take-all: spikes exist only at t=14 -> argmax of pot[14] over
// spiking (pot>0) positions, ties -> lowest flat index.
// ------------------------------------------------------------------
__device__ inline unsigned int orderedf(float f) {
  unsigned int u = __float_as_uint(f);
  return (u & 0x80000000u) ? ~u : (u | 0x80000000u);
}

__global__ void red_init_kernel(unsigned long long* red) {
  if (blockIdx.x == 0 && threadIdx.x == 0)
    *red = ((unsigned long long)orderedf(0.0f) << 32) | 0xFFFFFFFFull;
}

__global__ void winner_reduce_kernel(const float* __restrict__ pot14,
                                     unsigned long long* red, int n) {
  int i = blockIdx.x * blockDim.x + threadIdx.x;
  int stride = gridDim.x * blockDim.x;
  unsigned long long best = 0ull;
  for (; i < n; i += stride) {
    float v = pot14[i];
    if (v > 0.0f) {
      unsigned long long key =
          ((unsigned long long)orderedf(v) << 32) |
          (unsigned long long)(0xFFFFFFFFu - (unsigned int)i);
      if (key > best) best = key;
    }
  }
  if (best) atomicMax(red, best);
}

__global__ void winner_final_kernel(const unsigned long long* __restrict__ red,
                                    float* __restrict__ outClass) {
  if (blockIdx.x == 0 && threadIdx.x == 0) {
    unsigned long long key = *red;
    unsigned int fb  = (unsigned int)(key >> 32);
    unsigned int idx = 0xFFFFFFFFu - (unsigned int)(key & 0xFFFFFFFFull);
    bool hasSpike = fb > 0x80000000u;
    int feat = (int)(idx / (46u * 46u));
    *outClass = hasSpike ? (float)(feat / 60) : -1.0f;
  }
}

// ------------------------------------------------------------------
// host launcher
// ------------------------------------------------------------------
extern "C" void kernel_launch(void* const* d_in, const int* in_sizes, int n_in,
                              void* d_out, int out_size, void* d_ws, size_t ws_size,
                              hipStream_t stream) {
  (void)in_sizes; (void)n_in; (void)out_size; (void)ws_size;
  const float* in0 = (const float*)d_in[0];  // [15,6,256,256]
  const float* w1  = (const float*)d_in[1];  // [30,6,5,5]
  const float* w2  = (const float*)d_in[2];  // [250,30,3,3]
  const float* w31 = (const float*)d_in[3];  // [200,250,3,3]
  const float* w32 = (const float*)d_in[4];  // [200,250,5,5]
  const float* w33 = (const float*)d_in[5];  // [200,250,7,7]

  float* out = (float*)d_out;
  const size_t Ntot = (size_t)TSTEPS * 600 * 46 * 46;  // 19,044,000
  float* spkOut = out;
  float* potOut = out + Ntot;
  float* clsOut = out + 2 * Ntot;

  // workspace carve-up (~240 MB)
  char* ws = (char*)d_ws;
  size_t off = 0;
  auto alloc = [&](size_t bytes) -> void* {
    off = (off + 255) & ~(size_t)255;
    void* pp = (void*)(ws + off);
    off += bytes;
    return pp;
  };
  _Float16* x0h  = (_Float16*)alloc((size_t)TSTEPS * 6 * 256 * 256 * 2);
  _Float16* wr1  = (_Float16*)alloc((size_t)25 * 64 * 32 * 2);
  _Float16* wr2  = (_Float16*)alloc((size_t)9 * 256 * 32 * 2);
  _Float16* wr31 = (_Float16*)alloc((size_t)9 * 256 * 256 * 2);
  _Float16* wr32 = (_Float16*)alloc((size_t)25 * 256 * 256 * 2);
  _Float16* wr33 = (_Float16*)alloc((size_t)49 * 256 * 256 * 2);
  _Float16* spk1 = (_Float16*)alloc((size_t)TSTEPS * 30 * 256 * 256 * 2);
  _Float16* pl1  = (_Float16*)alloc((size_t)TSTEPS * 30 * 128 * 128 * 2);
  _Float16* spk2 = (_Float16*)alloc((size_t)TSTEPS * 250 * 128 * 128 * 2);
  _Float16* pl2  = (_Float16*)alloc((size_t)TSTEPS * 250 * 46 * 46 * 2);
  unsigned long long* red = (unsigned long long*)alloc(8);

  // 0) zero output (spk padding + t!=14 spikes + pot frame borders)
  zero_f32_kernel<<<2048, 256, 0, stream>>>(out, 2 * Ntot + 1);

  // 1) input f32->f16 + weight relayout to [KHKW][Fpad][Cpad]
  f32_to_f16_kernel<<<2048, 256, 0, stream>>>(in0, x0h,
                                              (size_t)TSTEPS * 6 * 256 * 256);
  relayout_w_kernel<<<64,   256, 0, stream>>>(w1,  wr1,  30,  6,  25, 64,  32);
  relayout_w_kernel<<<128,  256, 0, stream>>>(w2,  wr2,  250, 30, 9,  256, 32);
  relayout_w_kernel<<<512,  256, 0, stream>>>(w31, wr31, 200, 250, 9, 256, 256);
  relayout_w_kernel<<<1024, 256, 0, stream>>>(w32, wr32, 200, 250, 25, 256, 256);
  relayout_w_kernel<<<2048, 256, 0, stream>>>(w33, wr33, 200, 250, 49, 256, 256);

  // 2) layer 1: pad2 conv 5x5 (6->30) on 256x256, fire thr=15
  conv_wmma<6, 5, 5, 2, 0, 32, 64><<<dim3(4096, 1, TSTEPS), 128, 0, stream>>>(
      wr1, x0h, 30, 256, 256, 256, 256, 15.0f, spk1, nullptr, nullptr, 0, 0);
  pool2x2_kernel<<<2048, 256, 0, stream>>>(spk1, pl1, TSTEPS * 30, 256, 256);

  // 3) layer 2: pad1 conv 3x3 (30->250) on 128x128, fire thr=10
  conv_wmma<30, 3, 3, 1, 0, 32, 256><<<dim3(1024, 4, TSTEPS), 128, 0, stream>>>(
      wr2, pl1, 250, 128, 128, 128, 128, 10.0f, spk2, nullptr, nullptr, 0, 0);
  pool3pad2_kernel<<<2048, 256, 0, stream>>>(spk2, pl2, TSTEPS * 250);

  // 4) inception branches (250->200) into d_out frame
  red_init_kernel<<<1, 32, 0, stream>>>(red);
  conv_wmma<250, 3, 3, 0, 1, 256, 256><<<dim3(121, 4, TSTEPS), 128, 0, stream>>>(
      wr31, pl2, 200, 46, 46, 44, 44, 0.0f, nullptr, potOut, spkOut, 0, 1);
  conv_wmma<250, 5, 5, 0, 1, 256, 256><<<dim3(111, 4, TSTEPS), 128, 0, stream>>>(
      wr32, pl2, 200, 46, 46, 42, 42, 0.0f, nullptr, potOut, spkOut, 200, 2);
  conv_wmma<250, 7, 7, 0, 1, 256, 256><<<dim3(100, 4, TSTEPS), 128, 0, stream>>>(
      wr33, pl2, 200, 46, 46, 40, 40, 0.0f, nullptr, potOut, spkOut, 400, 3);

  // 5) winner-take-all over pot[t=14]
  winner_reduce_kernel<<<256, 256, 0, stream>>>(
      potOut + (size_t)(TSTEPS - 1) * 600 * 46 * 46, red, 600 * 46 * 46);
  winner_final_kernel<<<1, 32, 0, stream>>>(red, clsOut);
}
